// VideoColor_58188216926752
// MI455X (gfx1250) — matrix operand order repaired and checked by reading, not compiled, not run on hardware
//
#include <hip/hip_runtime.h>
#include <math.h>

// ---------------------------------------------------------------------------
// CDNA5 (gfx1250) implementation.
// All matrix math (3x3 convs via implicit GEMM, 1x1 proj conv, and the
// 9216x9216 correlation) runs on v_wmma_f32_16x16x32_bf16 (wave32).
// Correlation is fused with row-max/argmax (winner-take-all softmax at
// TEMP=1e-10) so the 340MB f-matrix is never materialized; theta/phi are
// kept in bf16 so tile staging is a raw copy (async-to-LDS when available).
// Workspace requirement: ~86 MB.
// ---------------------------------------------------------------------------

typedef __attribute__((ext_vector_type(16))) __bf16 v16bf;
typedef __attribute__((ext_vector_type(8)))  float  v8f;

union ABfrag { v16bf v; unsigned int u[8]; };

#if defined(__has_builtin)
#  if __has_builtin(__builtin_amdgcn_global_load_async_to_lds_b128) && \
      __has_builtin(__builtin_amdgcn_s_wait_asynccnt)
#    define USE_ASYNC_LDS 1
#  endif
#endif
#ifndef USE_ASYNC_LDS
#  define USE_ASYNC_LDS 0
#endif

#if USE_ASYNC_LDS
typedef int v4i_ __attribute__((vector_size(16)));
typedef v4i_ __attribute__((address_space(1))) gv4i;
typedef v4i_ __attribute__((address_space(3))) lv4i;
// Integer round-trip for address-space conversion:
//  - flat global addresses are numerically valid AS1 addresses
//  - flat LDS addresses keep the LDS byte offset in addr[31:0] (ISA 10.2)
__device__ inline void async_copy16(const void* g, void* l) {
  gv4i* gp = (gv4i*)(unsigned long long)g;
  lv4i* lp = (lv4i*)(unsigned int)(unsigned long long)l;
  __builtin_amdgcn_global_load_async_to_lds_b128(gp, lp, 0, 0);
}
#endif

__device__ inline unsigned short f2bf(float f) {
  unsigned int u = __float_as_uint(f);
  unsigned int r = (u + 0x7FFFu + ((u >> 16) & 1u)) >> 16;   // RNE
  return (unsigned short)r;
}

__device__ inline float block_reduce_sum(float v, float* sbuf) {
  int tid = threadIdx.x;
  sbuf[tid] = v; __syncthreads();
  for (int s = (int)blockDim.x >> 1; s > 0; s >>= 1) {
    if (tid < s) sbuf[tid] += sbuf[tid + s];
    __syncthreads();
  }
  float r = sbuf[0]; __syncthreads();
  return r;
}

// ---------------------------------------------------------------------------
// feature L2 normalization across channels (per pixel):  x / (||x||_C + 1e-6)
// ---------------------------------------------------------------------------
__global__ void featnorm_kernel(const float* __restrict__ in, float* __restrict__ out,
                                int C, int HW) {
  int p = blockIdx.x * blockDim.x + threadIdx.x;
  if (p >= HW) return;
  float s = 0.f;
  for (int c = 0; c < C; ++c) { float v = in[c * HW + p]; s += v * v; }
  float inv = 1.f / (sqrtf(s) + 1e-6f);
  for (int c = 0; c < C; ++c) out[c * HW + p] = in[c * HW + p] * inv;
}

// ---------------------------------------------------------------------------
// 3x3 conv, reflection pad 1, stride 1 or 2, implicit GEMM on WMMA bf16.
// Block: 128 threads = 4 waves; 32 cout x 64 pixels per block
// (each wave: 16 pixels, two 16-cout fragments -> 2 WMMAs per K-chunk).
// ---------------------------------------------------------------------------
__global__ __launch_bounds__(128)
void conv3x3_wmma(const float* __restrict__ in, const float* __restrict__ wgt,
                  const float* __restrict__ bias, float* __restrict__ out,
                  int Cin, int H, int W, int Cout, int OH, int OW, int stride) {
  const int K = Cin * 9;
  const int P = OH * OW;
  __shared__ __attribute__((aligned(16))) unsigned short As[32 * 32];
  __shared__ __attribute__((aligned(16))) unsigned short Bs[64 * 32];
  int tid  = threadIdx.x;
  int lane = tid & 31, wv = tid >> 5;
  int pixBase = blockIdx.x * 64;
  int coBase  = blockIdx.y * 32;
  v8f acc0 = {}, acc1 = {};

  for (int kc = 0; kc < K; kc += 32) {
    // stage weight tile [32 cout][32 k]
    for (int e = tid; e < 32 * 32; e += 128) {
      int r = e >> 5, kk = e & 31;
      int co = coBase + r, k = kc + kk;
      As[e] = f2bf((co < Cout && k < K) ? wgt[co * K + k] : 0.f);
    }
    // stage im2col tile [64 pix][32 k] with reflection padding
    for (int e = tid; e < 64 * 32; e += 128) {
      int pr = e >> 5, kk = e & 31;
      int p = pixBase + pr, k = kc + kk;
      float v = 0.f;
      if (p < P && k < K) {
        int ci = k / 9; int rs = k - ci * 9;
        int r3 = rs / 3; int s3 = rs - r3 * 3;
        int oy = p / OW, ox = p - oy * OW;
        int iy = oy * stride + r3 - 1;
        int ix = ox * stride + s3 - 1;
        iy = (iy < 0) ? -iy : iy; iy = (iy >= H) ? (2 * H - 2 - iy) : iy;
        ix = (ix < 0) ? -ix : ix; ix = (ix >= W) ? (2 * W - 2 - ix) : ix;
        v = in[ci * H * W + iy * W + ix];
      }
      Bs[e] = f2bf(v);
    }
    __syncthreads();

    ABfrag a0, a1, b;
    int mm = lane & 15, hb = lane >> 4;
    const unsigned int* A32 = (const unsigned int*)As;
    const unsigned int* B32 = (const unsigned int*)(Bs + wv * 512);
#pragma unroll
    for (int j = 0; j < 4; ++j) {
      int ko = 2 * j + 8 * hb;
      a0.u[j]     = A32[(mm * 32 + ko) >> 1];
      a1.u[j]     = A32[((mm + 16) * 32 + ko) >> 1];
      a0.u[4 + j] = A32[(mm * 32 + 16 + ko) >> 1];
      a1.u[4 + j] = A32[((mm + 16) * 32 + 16 + ko) >> 1];
    }
#pragma unroll
    for (int j = 0; j < 8; ++j) b.u[j] = B32[(mm * 32 + 2 * j + 8 * hb + ((j >> 2) << 3)) >> 1];
    acc0 = __builtin_amdgcn_wmma_f32_16x16x32_bf16(false, a0.v, false, b.v,
                                                   (short)0, acc0, false, false);
    acc1 = __builtin_amdgcn_wmma_f32_16x16x32_bf16(false, a1.v, false, b.v,
                                                   (short)0, acc1, false, false);
    __syncthreads();
  }

  int mm = lane & 15, hb = lane >> 4;
  int p = pixBase + wv * 16 + mm;
  if (p < P) {
#pragma unroll
    for (int r = 0; r < 8; ++r) {
      int co0 = coBase + r + 8 * hb;
      int co1 = co0 + 16;
      if (co0 < Cout) out[co0 * P + p] = acc0[r] + bias[co0];
      if (co1 < Cout) out[co1 * P + p] = acc1[r] + bias[co1];
    }
  }
}

// ---------------------------------------------------------------------------
// 1x1 conv (projection): out[cout][p] = W[cout][Cin] x in[Cin][p] + b
// Same 32x64 tiling as conv3x3.
// ---------------------------------------------------------------------------
__global__ __launch_bounds__(128)
void conv1x1_wmma(const float* __restrict__ in, const float* __restrict__ wgt,
                  const float* __restrict__ bias, float* __restrict__ out,
                  int Cin, int P, int Cout) {
  __shared__ __attribute__((aligned(16))) unsigned short As[32 * 32];
  __shared__ __attribute__((aligned(16))) unsigned short Bs[64 * 32];
  int tid  = threadIdx.x;
  int lane = tid & 31, wv = tid >> 5;
  int pixBase = blockIdx.x * 64;
  int coBase  = blockIdx.y * 32;
  v8f acc0 = {}, acc1 = {};

  for (int kc = 0; kc < Cin; kc += 32) {
    for (int e = tid; e < 32 * 32; e += 128) {
      int r = e >> 5, kk = e & 31;
      int co = coBase + r, k = kc + kk;
      As[e] = f2bf((co < Cout && k < Cin) ? wgt[co * Cin + k] : 0.f);
    }
    for (int e = tid; e < 64 * 32; e += 128) {
      int pr = e >> 5, kk = e & 31;
      int p = pixBase + pr, k = kc + kk;
      Bs[e] = f2bf((p < P && k < Cin) ? in[k * P + p] : 0.f);
    }
    __syncthreads();
    ABfrag a0, a1, b;
    int mm = lane & 15, hb = lane >> 4;
    const unsigned int* A32 = (const unsigned int*)As;
    const unsigned int* B32 = (const unsigned int*)(Bs + wv * 512);
#pragma unroll
    for (int j = 0; j < 4; ++j) {
      int ko = 2 * j + 8 * hb;
      a0.u[j]     = A32[(mm * 32 + ko) >> 1];
      a1.u[j]     = A32[((mm + 16) * 32 + ko) >> 1];
      a0.u[4 + j] = A32[(mm * 32 + 16 + ko) >> 1];
      a1.u[4 + j] = A32[((mm + 16) * 32 + 16 + ko) >> 1];
    }
#pragma unroll
    for (int j = 0; j < 8; ++j) b.u[j] = B32[(mm * 32 + 2 * j + 8 * hb + ((j >> 2) << 3)) >> 1];
    acc0 = __builtin_amdgcn_wmma_f32_16x16x32_bf16(false, a0.v, false, b.v,
                                                   (short)0, acc0, false, false);
    acc1 = __builtin_amdgcn_wmma_f32_16x16x32_bf16(false, a1.v, false, b.v,
                                                   (short)0, acc1, false, false);
    __syncthreads();
  }
  int mm = lane & 15, hb = lane >> 4;
  int p = pixBase + wv * 16 + mm;
  if (p < P) {
#pragma unroll
    for (int r = 0; r < 8; ++r) {
      int co0 = coBase + r + 8 * hb;
      int co1 = co0 + 16;
      if (co0 < Cout) out[co0 * P + p] = acc0[r] + bias[co0];
      if (co1 < Cout) out[co1 * P + p] = acc1[r] + bias[co1];
    }
  }
}

// ---------------------------------------------------------------------------
// instance-norm stats (per channel over HW) and in-place apply (+optional PReLU)
// ---------------------------------------------------------------------------
__global__ void inorm_stats_kernel(const float* __restrict__ x, float* __restrict__ mean,
                                   float* __restrict__ rstd, int HW) {
  __shared__ float sbuf[256];
  int c = blockIdx.x;
  const float* p = x + (long)c * HW;
  float s = 0.f, q = 0.f;
  for (int i = threadIdx.x; i < HW; i += (int)blockDim.x) {
    float v = p[i]; s += v; q += v * v;
  }
  float ts = block_reduce_sum(s, sbuf);
  float tq = block_reduce_sum(q, sbuf);
  if (threadIdx.x == 0) {
    float m = ts / (float)HW;
    float var = tq / (float)HW - m * m;
    mean[c] = m;
    rstd[c] = rsqrtf(var + 1e-5f);
  }
}

__global__ void inorm_apply_kernel(float* __restrict__ x, const float* __restrict__ mean,
                                   const float* __restrict__ rstd,
                                   const float* __restrict__ alpha,
                                   int HW, int n, int doPrelu) {
  int i = blockIdx.x * blockDim.x + threadIdx.x;
  if (i >= n) return;
  int c = i / HW;
  float v = (x[i] - mean[c]) * rstd[c];
  if (doPrelu && v < 0.f) v *= alpha[0];
  x[i] = v;
}

__global__ void add_prelu_kernel(float* __restrict__ x, const float* __restrict__ y,
                                 const float* __restrict__ alpha, int n) {
  int i = blockIdx.x * blockDim.x + threadIdx.x;
  if (i >= n) return;
  float v = x[i] + y[i];
  x[i] = (v >= 0.f) ? v : alpha[0] * v;
}

__global__ void up2_kernel(const float* __restrict__ in, float* __restrict__ out,
                           int C, int H, int W) {
  int n = C * 4 * H * W;
  int i = blockIdx.x * blockDim.x + threadIdx.x;
  if (i >= n) return;
  int OW = 2 * W;
  int c = i / (4 * H * W);
  int rem = i - c * 4 * H * W;
  int oy = rem / OW, ox = rem - oy * OW;
  out[i] = in[c * H * W + (oy >> 1) * W + (ox >> 1)];
}

// ---------------------------------------------------------------------------
// projection post-processing: per-channel spatial mean, then per-pixel channel
// L2 normalize; writes bf16 TRANSPOSED [n][256] so correlation staging is a
// raw contiguous copy.
// ---------------------------------------------------------------------------
__global__ void chan_mean_kernel(const float* __restrict__ t, float* __restrict__ cmean, int N) {
  __shared__ float sbuf[256];
  int c = blockIdx.x;
  float s = 0.f;
  for (int i = threadIdx.x; i < N; i += 256) s += t[(long)c * N + i];
  float ts = block_reduce_sum(s, sbuf);
  if (threadIdx.x == 0) cmean[c] = ts / (float)N;
}

__global__ void proj_norm_kernel(const float* __restrict__ t, const float* __restrict__ cmean,
                                 unsigned short* __restrict__ outT, int N) {
  __shared__ float sbuf[256];
  int n = blockIdx.x;
  int c = threadIdx.x;  // blockDim == 256 == C
  float v = t[(long)c * N + n] - cmean[c];
  float tot = block_reduce_sum(v * v, sbuf);
  float inv = 1.f / (sqrtf(tot) + 1e-6f);
  outT[(long)n * 256 + c] = f2bf(v * inv);
}

// ---------------------------------------------------------------------------
// fused correlation + winner-take-all.
// Block = 4 waves; wave w owns n-rows [blk*64 + w*16, +16) with its own theta
// block in LDS; all waves share each staged phi tile (async copy to LDS when
// the toolchain exposes the gfx1250 builtin). K = 256, 8 WMMAs per tile/wave.
// ---------------------------------------------------------------------------
__global__ __launch_bounds__(128)
void corr_argmax_kernel(const unsigned short* __restrict__ thetaT,
                        const unsigned short* __restrict__ phiT,
                        float* __restrict__ simmax, int* __restrict__ argmax,
                        int N, int M) {
  __shared__ __attribute__((aligned(16))) unsigned short Ash[4][16 * 256];
  __shared__ __attribute__((aligned(16))) unsigned short Bsh[16 * 256];
  __shared__ float redv[4][16][16];
  __shared__ int   redi[4][16][16];
  int tid = threadIdx.x, lane = tid & 31, wv = tid >> 5;
  int nBase = blockIdx.x * 64 + wv * 16;

  // stage this wave's theta block [16 n][256 c] (bf16, contiguous)
  {
    const uint4* src = (const uint4*)(thetaT + (long)nBase * 256);
    uint4* dst = (uint4*)&Ash[wv][0];
    for (int e = lane; e < 512; e += 32) dst[e] = src[e];
  }

  float best[8]; int bidx[8];
#pragma unroll
  for (int r = 0; r < 8; ++r) { best[r] = -3.4e38f; bidx[r] = 0; }
  int nn = lane & 15, hb = lane >> 4;
  const unsigned int* A32 = (const unsigned int*)&Ash[wv][0];
  const unsigned int* B32 = (const unsigned int*)Bsh;
  __syncthreads();

  for (int mt = 0; mt < M; mt += 16) {
    // stage phi tile [16 m][256 c] cooperatively: 8KB contiguous
    const unsigned short* src = phiT + (long)mt * 256;
#if USE_ASYNC_LDS
    for (int e = tid; e < 512; e += 128)
      async_copy16(src + (long)e * 8, Bsh + e * 8);
    __builtin_amdgcn_s_wait_asynccnt(0);
#else
    {
      const uint4* s4 = (const uint4*)src;
      uint4* d4 = (uint4*)Bsh;
      for (int e = tid; e < 512; e += 128) d4[e] = s4[e];
    }
#endif
    if (mt + 16 < M)  // prefetch next tile (global_prefetch_b8)
      __builtin_prefetch((const char*)(phiT + (long)(mt + 16) * 256) + tid * 64, 0, 0);
    __syncthreads();

    v8f acc = {};
#pragma unroll
    for (int kc = 0; kc < 256; kc += 32) {
      ABfrag a, b;
#pragma unroll
      for (int j = 0; j < 4; ++j) {
        int ko = kc + 2 * j + 8 * hb;
        a.u[j]     = A32[(nn * 256 + ko) >> 1];
        a.u[4 + j] = A32[(nn * 256 + 16 + ko) >> 1];
      }
#pragma unroll
      for (int j = 0; j < 8; ++j) b.u[j] = B32[(nn * 256 + kc + 2 * j + 8 * hb + ((j >> 2) << 3)) >> 1];
      acc = __builtin_amdgcn_wmma_f32_16x16x32_bf16(false, a.v, false, b.v,
                                                    (short)0, acc, false, false);
    }
    int mcol = mt + nn;  // this lane's column
#pragma unroll
    for (int r = 0; r < 8; ++r)
      if (acc[r] > best[r]) { best[r] = acc[r]; bidx[r] = mcol; }
    __syncthreads();
  }

#pragma unroll
  for (int r = 0; r < 8; ++r) {
    redv[wv][r + 8 * hb][nn] = best[r];
    redi[wv][r + 8 * hb][nn] = bidx[r];
  }
  __syncthreads();
  if (lane < 16) {
    float bv = redv[wv][lane][0]; int bi = redi[wv][lane][0];
    for (int j = 1; j < 16; ++j)
      if (redv[wv][lane][j] > bv) { bv = redv[wv][lane][j]; bi = redi[wv][lane][j]; }
    simmax[nBase + lane] = bv;
    argmax[nBase + lane] = bi;
  }
}

// avg_pool2d(IB_lab, 4): [3,384,384] -> [3,96*96]
__global__ void avgpool4_kernel(const float* __restrict__ ib, float* __restrict__ blab) {
  int i = blockIdx.x * blockDim.x + threadIdx.x;
  if (i >= 3 * 9216) return;
  int c = i / 9216, n = i - c * 9216;
  int oy = n / 96, ox = n - oy * 96;
  float s = 0.f;
  for (int dy = 0; dy < 4; ++dy)
    for (int dx = 0; dx < 4; ++dx)
      s += ib[c * 147456 + (oy * 4 + dy) * 384 + (ox * 4 + dx)];
  blab[i] = s * (1.f / 16.f);
}

// gather via argmax + 4x upsample of both outputs into d_out
__global__ void output_kernel(const float* __restrict__ blab, const float* __restrict__ simmax,
                              const int* __restrict__ argmax, float* __restrict__ out) {
  int i = blockIdx.x * blockDim.x + threadIdx.x;
  if (i >= 4 * 147456) return;
  int c = i / 147456, rem = i - c * 147456;
  int y = rem / 384, x = rem - y * 384;
  int n = (y >> 2) * 96 + (x >> 2);
  if (c < 3) out[c * 147456 + rem] = blab[c * 9216 + argmax[n]];
  else       out[3 * 147456 + rem] = simmax[n];
}

// ---------------------------------------------------------------------------
// host orchestration
// ---------------------------------------------------------------------------
enum {
  IN_IB = 0, IN_AF2, IN_AF3, IN_AF4, IN_AF5, IN_BF2, IN_BF3, IN_BF4, IN_BF5,
  P_L2W1, P_L2B1, P_L2A1, P_L2W2, P_L2B2, P_L2A2,
  P_L3W1, P_L3B1, P_L3A1, P_L3W2, P_L3B2, P_L3A2,
  P_L4W1, P_L4B1, P_L4A1, P_L4W2, P_L4B2, P_L4A2,
  P_L5W1, P_L5B1, P_L5A1, P_L5W2, P_L5B2, P_L5A2,
  P_RB0W1, P_RB0B1, P_RB0W2, P_RB0B2, P_RB0A,
  P_RB1W1, P_RB1B1, P_RB1W2, P_RB1B2, P_RB1A,
  P_RB2W1, P_RB2B1, P_RB2W2, P_RB2B2, P_RB2A,
  P_THW, P_THB, P_PHW, P_PHB, N_IN_TOTAL
};

extern "C" void kernel_launch(void* const* d_in, const int* in_sizes, int n_in,
                              void* d_out, int out_size, void* d_ws, size_t ws_size,
                              hipStream_t stream) {
  (void)in_sizes; (void)out_size; (void)ws_size;
  const float* PP[N_IN_TOTAL];
  for (int i = 0; i < N_IN_TOTAL && i < n_in; ++i) PP[i] = (const float*)d_in[i];

  float* ws = (float*)d_ws;
  float* FN    = ws;                       // 4,718,592  (128*192*192)
  float* T1    = ws + 4718592ull;          // 4,718,592
  float* T2    = ws + 9437184ull;          // 2,359,296
  float* FEATS = ws + 11796480ull;         // 2,359,296  (256*96*96)
  unsigned short* THETA = (unsigned short*)(ws + 14155776ull);  // bf16 [n][256]
  unsigned short* PHI   = (unsigned short*)(ws + 16515072ull);  // bf16 [m][256]
  float* PTMP  = ws + 18874368ull;         // 2,359,296  proj conv out [256][9216]
  float* MEANB = ws + 21233664ull;         // 256
  float* RSTDB = MEANB + 256;              // 256
  float* CMEAN = RSTDB + 256;              // 256
  float* SIMMX = CMEAN + 256;              // 9216
  float* BLAB  = SIMMX + 9216;             // 27648
  int*   ARGMX = (int*)(BLAB + 27648);     // 9216 ints

  auto conv3 = [&](const float* in, int Cin, int H, int W,
                   const float* w, const float* b, float* out, int Cout, int stride) {
    int OH = (H - 1) / stride + 1;   // pad=1, k=3
    int OW = (W - 1) / stride + 1;
    dim3 grid((OH * OW + 63) / 64, (Cout + 31) / 32);
    conv3x3_wmma<<<grid, 128, 0, stream>>>(in, w, b, out, Cin, H, W, Cout, OH, OW, stride);
  };
  auto inormA = [&](float* x, int C, int HW, const float* alpha, int doPrelu) {
    inorm_stats_kernel<<<C, 256, 0, stream>>>(x, MEANB, RSTDB, HW);
    int n = C * HW;
    inorm_apply_kernel<<<(n + 255) / 256, 256, 0, stream>>>(x, MEANB, RSTDB, alpha, HW, n, doPrelu);
  };

  auto embed = [&](const float* f2, const float* f3, const float* f4, const float* f5) {
    // branch2 -> feats[0:64]
    featnorm_kernel<<<(36864 + 255) / 256, 256, 0, stream>>>(f2, FN, 128, 36864);
    conv3(FN, 128, 192, 192, PP[P_L2W1], PP[P_L2B1], T1, 128, 1);
    inormA(T1, 128, 36864, PP[P_L2A1], 1);
    conv3(T1, 128, 192, 192, PP[P_L2W2], PP[P_L2B2], FEATS, 64, 2);
    inormA(FEATS, 64, 9216, PP[P_L2A2], 1);
    // branch3 -> feats[64:128]
    featnorm_kernel<<<(9216 + 255) / 256, 256, 0, stream>>>(f3, FN, 256, 9216);
    conv3(FN, 256, 96, 96, PP[P_L3W1], PP[P_L3B1], T1, 128, 1);
    inormA(T1, 128, 9216, PP[P_L3A1], 1);
    conv3(T1, 128, 96, 96, PP[P_L3W2], PP[P_L3B2], FEATS + 64 * 9216, 64, 1);
    inormA(FEATS + 64 * 9216, 64, 9216, PP[P_L3A2], 1);
    // branch4 -> feats[128:192] (up2 after cbp chain)
    featnorm_kernel<<<(2304 + 255) / 256, 256, 0, stream>>>(f4, FN, 512, 2304);
    conv3(FN, 512, 48, 48, PP[P_L4W1], PP[P_L4B1], T1, 256, 1);
    inormA(T1, 256, 2304, PP[P_L4A1], 1);
    conv3(T1, 256, 48, 48, PP[P_L4W2], PP[P_L4B2], T2, 64, 1);
    inormA(T2, 64, 2304, PP[P_L4A2], 1);
    up2_kernel<<<(64 * 9216 + 255) / 256, 256, 0, stream>>>(T2, FEATS + 128 * 9216, 64, 48, 48);
    // branch5 -> feats[192:256] (up2(cbp2(up2(cbp1(x)))))
    featnorm_kernel<<<(576 + 255) / 256, 256, 0, stream>>>(f5, FN, 512, 576);
    conv3(FN, 512, 24, 24, PP[P_L5W1], PP[P_L5B1], T1, 256, 1);
    inormA(T1, 256, 576, PP[P_L5A1], 1);
    up2_kernel<<<(256 * 2304 + 255) / 256, 256, 0, stream>>>(T1, T2, 256, 24, 24);
    conv3(T2, 256, 48, 48, PP[P_L5W2], PP[P_L5B2], T1, 64, 1);
    inormA(T1, 64, 2304, PP[P_L5A2], 1);
    up2_kernel<<<(64 * 9216 + 255) / 256, 256, 0, stream>>>(T1, FEATS + 192 * 9216, 64, 48, 48);
    // trunk: 3 resblocks
    const int RW1[3] = {P_RB0W1, P_RB1W1, P_RB2W1};
    const int RBB1[3] = {P_RB0B1, P_RB1B1, P_RB2B1};
    const int RW2[3] = {P_RB0W2, P_RB1W2, P_RB2W2};
    const int RBB2[3] = {P_RB0B2, P_RB1B2, P_RB2B2};
    const int RA[3]  = {P_RB0A, P_RB1A, P_RB2A};
    for (int i = 0; i < 3; ++i) {
      conv3(FEATS, 256, 96, 96, PP[RW1[i]], PP[RBB1[i]], T1, 256, 1);
      inormA(T1, 256, 9216, PP[RA[i]], 1);
      conv3(T1, 256, 96, 96, PP[RW2[i]], PP[RBB2[i]], T2, 256, 1);
      inormA(T2, 256, 9216, PP[RA[i]], 0);
      add_prelu_kernel<<<(256 * 9216 + 255) / 256, 256, 0, stream>>>(FEATS, T2, PP[RA[i]], 256 * 9216);
    }
  };

  auto proj = [&](const float* pw, const float* pb, unsigned short* outT) {
    dim3 g((9216 + 63) / 64, (256 + 31) / 32);
    conv1x1_wmma<<<g, 128, 0, stream>>>(FEATS, pw, pb, PTMP, 256, 9216, 256);
    chan_mean_kernel<<<256, 256, 0, stream>>>(PTMP, CMEAN, 9216);
    proj_norm_kernel<<<9216, 256, 0, stream>>>(PTMP, CMEAN, outT, 9216);
  };

  // image A -> theta ; image B -> phi (shared temps, sequential)
  embed(PP[IN_AF2], PP[IN_AF3], PP[IN_AF4], PP[IN_AF5]);
  proj(PP[P_THW], PP[P_THB], THETA);
  embed(PP[IN_BF2], PP[IN_BF3], PP[IN_BF4], PP[IN_BF5]);
  proj(PP[P_PHW], PP[P_PHB], PHI);

  // pooled Lab, fused correlation/argmax, gather + 4x upsample
  avgpool4_kernel<<<(3 * 9216 + 255) / 256, 256, 0, stream>>>(PP[IN_IB], BLAB);
  corr_argmax_kernel<<<9216 / 64, 128, 0, stream>>>(THETA, PHI, SIMMX, ARGMX, 9216, 9216);
  output_kernel<<<(4 * 147456 + 255) / 256, 256, 0, stream>>>(BLAB, SIMMX, ARGMX, (float*)d_out);
}
